// AgentFormerSceneDecoder_61134564491679
// MI455X (gfx1250) — compile-verified, matrix operand order checked by reading, not run on hardware
//
#include <hip/hip_runtime.h>
#include <hip/hip_bf16.h>
#include <math.h>

typedef _Float16 f16;
typedef __attribute__((ext_vector_type(16))) _Float16 v16h;
typedef __attribute__((ext_vector_type(8)))  float    v8f;

#define A_N   128   // agents
#define D_N   256   // embedding dim
#define H_N   8     // heads
#define DH    32    // head dim
#define MLP_N 1024
#define MEM_N 1024  // memory tokens
#define T_N   12    // pred len
#define HD    128   // decoder_h_dim
#define KPAD  160   // 2+HD=130 padded to multiple of 32
#define CTXMAX (T_N * A_N)  // 1536

// ---------------------------------------------------------------------------
// WMMA fragment loader: 16x32 f16 tile slice for one lane.
// A is row-major [M][K] (lda in elems); W is [N][K] row-major, same pattern.
// Lane layout (CDNA5 16-bit A 16x32): lanes 0-15 -> M=lane, K 0-7 & 16-23;
// lanes 16-31 -> M=lane-16, K 8-15 & 24-31.  Two 16-byte loads per lane.
// ---------------------------------------------------------------------------
__device__ __forceinline__ v16h frag_load(const f16* tile, int ld, int k0, int lane) {
    int r = lane & 15;
    int h = lane >> 4;
    const f16* p = tile + (long)r * ld + k0 + h * 8;
    v16h out;
    ((uint4*)&out)[0] = *(const uint4*)p;         // elems 0..7   (K = h*8 + 0..7)
    ((uint4*)&out)[1] = *(const uint4*)(p + 16);  // elems 8..15  (K = 16 + h*8 + 0..7)
    return out;
}

// ---------------------------------------------------------------------------
// Register-blocked WMMA GEMM, f32 output.
//   C[m][n] = scale * sum_k A[m][k]*W[n][k] + bias[n]
// Block = 128 threads = 4 waves; each wave owns one 16-row M tile and an
// NT*16-wide N strip: the A fragment is loaded once per K chunk and reused
// across NT WMMAs (4x less A traffic, 16x more work per block).
// blockIdx.z selects a slice (attention head) via element strides aZ/wZ/cZ.
// ---------------------------------------------------------------------------
template <int NT>
__global__ void k_gemm_f32(const f16* __restrict__ Am, int lda, long aZ,
                           const f16* __restrict__ Wm, int ldw, long wZ,
                           const float* __restrict__ bias,
                           float* __restrict__ Cm, int ldc, long cZ,
                           int K, float scale) {
    int lane = threadIdx.x & 31;
    int wave = threadIdx.x >> 5;
    int tn0 = blockIdx.x * NT;
    int tm  = blockIdx.y * 4 + wave;
    int z   = blockIdx.z;
    const f16* Ab = Am + (long)z * aZ + (long)tm * 16 * lda;
    const f16* Wb = Wm + (long)z * wZ + (long)tn0 * 16 * ldw;
    v8f zero = {0.f, 0.f, 0.f, 0.f, 0.f, 0.f, 0.f, 0.f};
    v8f acc[NT];
#pragma unroll
    for (int t = 0; t < NT; ++t) acc[t] = zero;
    for (int k0 = 0; k0 < K; k0 += 32) {
        v16h a = frag_load(Ab, lda, k0, lane);
#pragma unroll
        for (int t = 0; t < NT; ++t) {
            v16h b = frag_load(Wb + (long)t * 16 * ldw, ldw, k0, lane);
            acc[t] = __builtin_amdgcn_wmma_f32_16x16x32_f16(false, a, false, b,
                                                            (short)0, acc[t], false, false);
        }
    }
    int c = lane & 15, h = lane >> 4;
#pragma unroll
    for (int t = 0; t < NT; ++t) {
        int n = (tn0 + t) * 16 + c;
        float bv = bias ? bias[n] : 0.f;
        float* Cb = Cm + (long)z * cZ + (long)(tm * 16) * ldc + n;
#pragma unroll
        for (int v = 0; v < 8; ++v)
            Cb[(long)(v + 8 * h) * ldc] = acc[t][v] * scale + bv;
    }
}

// ---------------------------------------------------------------------------
// Register-blocked WMMA GEMM, f16 output(s): optional row-major out and/or
// transposed out (for V caches stored [D][ctx]); optional ReLU (FFN layer 1).
// ---------------------------------------------------------------------------
template <int NT>
__global__ void k_gemm_f16(const f16* __restrict__ Am, int lda, long aZ,
                           const f16* __restrict__ Wm, int ldw, long wZ,
                           const float* __restrict__ bias,
                           f16* __restrict__ outRM, int ldo, long oZ,
                           f16* __restrict__ outT, int ldT, long tZ,
                           int K, int relu) {
    int lane = threadIdx.x & 31;
    int wave = threadIdx.x >> 5;
    int tn0 = blockIdx.x * NT;
    int tm  = blockIdx.y * 4 + wave;
    int z   = blockIdx.z;
    const f16* Ab = Am + (long)z * aZ + (long)tm * 16 * lda;
    const f16* Wb = Wm + (long)z * wZ + (long)tn0 * 16 * ldw;
    v8f zero = {0.f, 0.f, 0.f, 0.f, 0.f, 0.f, 0.f, 0.f};
    v8f acc[NT];
#pragma unroll
    for (int t = 0; t < NT; ++t) acc[t] = zero;
    for (int k0 = 0; k0 < K; k0 += 32) {
        v16h a = frag_load(Ab, lda, k0, lane);
#pragma unroll
        for (int t = 0; t < NT; ++t) {
            v16h b = frag_load(Wb + (long)t * 16 * ldw, ldw, k0, lane);
            acc[t] = __builtin_amdgcn_wmma_f32_16x16x32_f16(false, a, false, b,
                                                            (short)0, acc[t], false, false);
        }
    }
    int c = lane & 15, h = lane >> 4;
#pragma unroll
    for (int t = 0; t < NT; ++t) {
        int n = (tn0 + t) * 16 + c;
        float bv = bias ? bias[n] : 0.f;
#pragma unroll
        for (int v = 0; v < 8; ++v) {
            float val = acc[t][v] + bv;
            if (relu) val = fmaxf(val, 0.f);
            int m = tm * 16 + v + 8 * h;
            if (outRM) outRM[(long)z * oZ + (long)m * ldo + n] = (f16)val;
            if (outT)  outT [(long)z * tZ + (long)n * ldT + m] = (f16)val;
        }
    }
}

// ---------------------------------------------------------------------------
// Masked softmax over one (row, head) pair; writes f16 probabilities.
// mask[i][j] = agent_mask[i][j mod 128] (block-causal part is implicit:
// new rows have max time so the whole cache is visible).
// ---------------------------------------------------------------------------
__global__ void k_softmax(const float* __restrict__ S,
                          const float* __restrict__ amask,
                          f16* __restrict__ P, int ctx) {
    int r = blockIdx.x, head = blockIdx.y, t = threadIdx.x;
    const float* srow = S + ((long)head * A_N + r) * ctx;
    f16* prow = P + ((long)head * A_N + r) * ctx;
    const float* mrow = amask + (long)r * A_N;
    __shared__ float red[256];
    float mx = -3.0e38f;
    for (int j = t; j < ctx; j += 256)
        mx = fmaxf(mx, srow[j] + mrow[j & (A_N - 1)]);
    red[t] = mx; __syncthreads();
    for (int s = 128; s > 0; s >>= 1) { if (t < s) red[t] = fmaxf(red[t], red[t + s]); __syncthreads(); }
    mx = red[0]; __syncthreads();
    float sum = 0.f;
    for (int j = t; j < ctx; j += 256)
        sum += __expf(srow[j] + mrow[j & (A_N - 1)] - mx);
    red[t] = sum; __syncthreads();
    for (int s = 128; s > 0; s >>= 1) { if (t < s) red[t] += red[t + s]; __syncthreads(); }
    float inv = 1.f / red[0];
    for (int j = t; j < ctx; j += 256)
        prow[j] = (f16)(__expf(srow[j] + mrow[j & (A_N - 1)] - mx) * inv);
}

// x = LayerNorm(x + add) * g + b ; also emit f16 copy for next WMMA GEMM.
__global__ void k_ln(float* __restrict__ x, const float* __restrict__ add,
                     const float* __restrict__ g, const float* __restrict__ b,
                     f16* __restrict__ x16) {
    int r = blockIdx.x, c = threadIdx.x;  // 256 threads
    __shared__ float red[256];
    float v = x[r * D_N + c] + add[r * D_N + c];
    red[c] = v; __syncthreads();
    for (int s = 128; s > 0; s >>= 1) { if (c < s) red[c] += red[c + s]; __syncthreads(); }
    float mean = red[0] * (1.f / D_N); __syncthreads();
    float d = v - mean;
    red[c] = d * d; __syncthreads();
    for (int s = 128; s > 0; s >>= 1) { if (c < s) red[c] += red[c + s]; __syncthreads(); }
    float var = red[0] * (1.f / D_N);
    float y = d * rsqrtf(var + 1e-5f) * g[c] + b[c];
    x[r * D_N + c] = y;
    x16[r * D_N + c] = (f16)y;
}

// Add time-PE(step) + agent-PE(row) to token embedding; emit f16 copy.
__global__ void k_addpe(float* __restrict__ x, f16* __restrict__ x16, int s) {
    int r = blockIdx.x, c = threadIdx.x;
    float di = __expf(-logf(10000.f) * (float)(c & ~1) / (float)D_N);
    float pt = (c & 1) ? __cosf((float)s * di) : __sinf((float)s * di);
    float pa = (c & 1) ? __cosf((float)r * di) : __sinf((float)r * di);
    float v = x[r * D_N + c] + pt + pa;
    x[r * D_N + c] = v;
    x16[r * D_N + c] = (f16)v;
}

// feat = [scene_pos, decoder_state, zero-pad] as f16 [128][160]
__global__ void k_feat(const float* __restrict__ sp, const float* __restrict__ ds,
                       f16* __restrict__ feat) {
    int r = blockIdx.x, c = threadIdx.x;  // 160 threads
    float v = 0.f;
    if (c < 2)        v = sp[r * 2 + c];
    else if (c < 130) v = ds[r * HD + (c - 2)];
    feat[r * KPAD + c] = (f16)v;
}

// rel = x @ out_W.T + out_b ; scene_pos += rel ; write step output
__global__ void k_outproj(const float* __restrict__ x, const float* __restrict__ oW,
                          const float* __restrict__ ob, float* __restrict__ sp,
                          float* __restrict__ outbuf) {
    int r = threadIdx.x;  // 128
    float a0 = ob[0], a1 = ob[1];
    for (int k = 0; k < D_N; ++k) {
        float xv = x[r * D_N + k];
        a0 += xv * oW[k];
        a1 += xv * oW[D_N + k];
    }
    outbuf[r * 2 + 0] = a0;
    outbuf[r * 2 + 1] = a1;
    sp[r * 2 + 0] += a0;
    sp[r * 2 + 1] += a1;
}

__global__ void k_cvt(const float* __restrict__ s, f16* __restrict__ d, int n) {
    int i = blockIdx.x * 256 + threadIdx.x;
    if (i < n) d[i] = (f16)s[i];
}

__global__ void k_cvt_pad(const float* __restrict__ s, f16* __restrict__ d,
                          int rows, int kin, int kout) {
    int i = blockIdx.x * 256 + threadIdx.x;
    if (i >= rows * kout) return;
    int r = i / kout, c = i % kout;
    d[i] = (c < kin) ? (f16)s[r * kin + c] : (f16)0.f;
}

__global__ void k_init_sp(const float* __restrict__ lp, float* __restrict__ sp) {
    int i = threadIdx.x;  // 256
    sp[i] = lp[i];
}

// ---------------------------------------------------------------------------
extern "C" void kernel_launch(void* const* d_in, const int* in_sizes, int n_in,
                              void* d_out, int out_size, void* d_ws, size_t ws_size,
                              hipStream_t stream) {
    (void)in_sizes; (void)n_in; (void)out_size; (void)ws_size;

    const float* last_pos  = (const float*)d_in[0];
    const float* dec_state = (const float*)d_in[1];
    const float* memory    = (const float*)d_in[2];
    const float* amask     = (const float*)d_in[3];
    const float* in_W      = (const float*)d_in[4];
    const float* in_b      = (const float*)d_in[5];
    const float* out_W     = (const float*)d_in[6];
    const float* out_b     = (const float*)d_in[7];
    const float* sa_Wq = (const float*)d_in[8],  *sa_bq = (const float*)d_in[9];
    const float* sa_Wk = (const float*)d_in[10], *sa_bk = (const float*)d_in[11];
    const float* sa_Wv = (const float*)d_in[12], *sa_bv = (const float*)d_in[13];
    const float* sa_Wo = (const float*)d_in[14], *sa_bo = (const float*)d_in[15];
    const float* ca_Wq = (const float*)d_in[16], *ca_bq = (const float*)d_in[17];
    const float* ca_Wk = (const float*)d_in[18], *ca_bk = (const float*)d_in[19];
    const float* ca_Wv = (const float*)d_in[20], *ca_bv = (const float*)d_in[21];
    const float* ca_Wo = (const float*)d_in[22], *ca_bo = (const float*)d_in[23];
    const float* ff_W1 = (const float*)d_in[24], *ff_b1 = (const float*)d_in[25];
    const float* ff_W2 = (const float*)d_in[26], *ff_b2 = (const float*)d_in[27];
    const float* ln1_g = (const float*)d_in[28], *ln1_b = (const float*)d_in[29];
    const float* ln2_g = (const float*)d_in[30], *ln2_b = (const float*)d_in[31];
    const float* ln3_g = (const float*)d_in[32], *ln3_b = (const float*)d_in[33];
    float* outp = (float*)d_out;  // [12][128][2] f32

    // ---- workspace carve-up (256B aligned slabs) ----
    size_t off = 0;
    auto alloc_h = [&](size_t cnt) -> f16* {
        off = (off + 255) & ~(size_t)255;
        f16* p = (f16*)((char*)d_ws + off);
        off += cnt * sizeof(f16);
        return p;
    };
    auto alloc_f = [&](size_t cnt) -> float* {
        off = (off + 255) & ~(size_t)255;
        float* p = (float*)((char*)d_ws + off);
        off += cnt * sizeof(float);
        return p;
    };
    const size_t WSZ = 2 * D_N * D_N;  // per [2][256][256] weight array
    f16* saWq16 = alloc_h(WSZ); f16* saWk16 = alloc_h(WSZ);
    f16* saWv16 = alloc_h(WSZ); f16* saWo16 = alloc_h(WSZ);
    f16* caWq16 = alloc_h(WSZ); f16* caWk16 = alloc_h(WSZ);
    f16* caWv16 = alloc_h(WSZ); f16* caWo16 = alloc_h(WSZ);
    f16* ffW1_16 = alloc_h(2 * MLP_N * D_N);
    f16* ffW2_16 = alloc_h(2 * D_N * MLP_N);
    f16* inW16   = alloc_h(D_N * KPAD);
    f16* mem16   = alloc_h(MEM_N * D_N);
    f16* Kmem16  = alloc_h(2 * MEM_N * D_N);   // [l][1024][256]
    f16* VmemT16 = alloc_h(2 * D_N * MEM_N);   // [l][256][1024]
    f16* Kc16    = alloc_h(2 * CTXMAX * D_N);  // [l][1536][256]  grows by step
    f16* VcT16   = alloc_h(2 * D_N * CTXMAX);  // [l][256][1536]
    f16* feat16  = alloc_h(A_N * KPAD);
    f16* Q16     = alloc_h(A_N * D_N);
    f16* O16     = alloc_h(A_N * D_N);
    f16* H116    = alloc_h(A_N * MLP_N);
    f16* x16     = alloc_h(A_N * D_N);
    f16* P16     = alloc_h((size_t)H_N * A_N * CTXMAX);
    float* Sf   = alloc_f((size_t)H_N * A_N * CTXMAX);
    float* xf   = alloc_f(A_N * D_N);
    float* tmpf = alloc_f(A_N * D_N);
    float* sp   = alloc_f(A_N * 2);

    const float ISQ = 0.17677669529663687f;  // 1/sqrt(32)

    // ---- one-time (per call) weight conversion to f16 ----
    struct { const float* s; f16* d; int n; } cv[] = {
        {sa_Wq, saWq16, (int)WSZ}, {sa_Wk, saWk16, (int)WSZ},
        {sa_Wv, saWv16, (int)WSZ}, {sa_Wo, saWo16, (int)WSZ},
        {ca_Wq, caWq16, (int)WSZ}, {ca_Wk, caWk16, (int)WSZ},
        {ca_Wv, caWv16, (int)WSZ}, {ca_Wo, caWo16, (int)WSZ},
        {ff_W1, ffW1_16, 2 * MLP_N * D_N}, {ff_W2, ffW2_16, 2 * D_N * MLP_N},
        {memory, mem16, MEM_N * D_N},
    };
    for (auto& c : cv)
        k_cvt<<<(c.n + 255) / 256, 256, 0, stream>>>(c.s, c.d, c.n);
    k_cvt_pad<<<(D_N * KPAD + 255) / 256, 256, 0, stream>>>(in_W, inW16, D_N, 2 + HD, KPAD);
    k_init_sp<<<1, 256, 0, stream>>>(last_pos, sp);

    // ---- precompute cross-attention K/V of the fixed memory, per layer ----
    // M=1024 -> grid.y=16 (4 waves/block), N=256 -> grid.x=4 (NT=4)
    for (int l = 0; l < 2; ++l) {
        k_gemm_f16<4><<<dim3(4, 16, 1), 128, 0, stream>>>(
            mem16, D_N, 0, caWk16 + (size_t)l * D_N * D_N, D_N, 0, ca_bk + l * D_N,
            Kmem16 + (size_t)l * MEM_N * D_N, D_N, 0, nullptr, 0, 0, D_N, 0);
        k_gemm_f16<4><<<dim3(4, 16, 1), 128, 0, stream>>>(
            mem16, D_N, 0, caWv16 + (size_t)l * D_N * D_N, D_N, 0, ca_bv + l * D_N,
            nullptr, 0, 0, VmemT16 + (size_t)l * D_N * MEM_N, MEM_N, 0, D_N, 0);
    }

    // ---- autoregressive loop with self-attn KV cache ----
    for (int s = 0; s < T_N; ++s) {
        int ctx = (s + 1) * A_N;
        // token embedding for the 128 new rows
        k_feat<<<A_N, KPAD, 0, stream>>>(sp, dec_state, feat16);
        k_gemm_f32<4><<<dim3(4, 2, 1), 128, 0, stream>>>(
            feat16, KPAD, 0, inW16, KPAD, 0, in_b, xf, D_N, 0, KPAD, 1.f);
        k_addpe<<<A_N, D_N, 0, stream>>>(xf, x16, s);

        for (int l = 0; l < 2; ++l) {
            const f16* Wq = saWq16 + (size_t)l * D_N * D_N;
            const f16* Wk = saWk16 + (size_t)l * D_N * D_N;
            const f16* Wv = saWv16 + (size_t)l * D_N * D_N;
            const f16* Wo = saWo16 + (size_t)l * D_N * D_N;
            f16* Kc  = Kc16  + (size_t)l * CTXMAX * D_N;
            f16* VcT = VcT16 + (size_t)l * D_N * CTXMAX;

            // ---- self-attention ----
            k_gemm_f16<4><<<dim3(4, 2, 1), 128, 0, stream>>>(
                x16, D_N, 0, Wq, D_N, 0, sa_bq + l * D_N,
                Q16, D_N, 0, nullptr, 0, 0, D_N, 0);
            k_gemm_f16<4><<<dim3(4, 2, 1), 128, 0, stream>>>(        // append K rows
                x16, D_N, 0, Wk, D_N, 0, sa_bk + l * D_N,
                Kc + (size_t)s * A_N * D_N, D_N, 0, nullptr, 0, 0, D_N, 0);
            k_gemm_f16<4><<<dim3(4, 2, 1), 128, 0, stream>>>(        // append V cols (transposed)
                x16, D_N, 0, Wv, D_N, 0, sa_bv + l * D_N,
                nullptr, 0, 0, VcT + s * A_N, CTXMAX, 0, D_N, 0);
            // per-head scores: K=dh=32 -> exactly one WMMA chunk per tile
            k_gemm_f32<4><<<dim3(ctx / 64, 2, H_N), 128, 0, stream>>>(
                Q16, D_N, DH, Kc, D_N, DH, nullptr,
                Sf, ctx, (long)A_N * ctx, DH, ISQ);
            k_softmax<<<dim3(A_N, H_N), 256, 0, stream>>>(Sf, amask, P16, ctx);
            // per-head P @ V  (N=32 per head -> NT=2)
            k_gemm_f16<2><<<dim3(1, 2, H_N), 128, 0, stream>>>(
                P16, ctx, (long)A_N * ctx, VcT, CTXMAX, (long)DH * CTXMAX, nullptr,
                O16, D_N, DH, nullptr, 0, 0, ctx, 0);
            k_gemm_f32<4><<<dim3(4, 2, 1), 128, 0, stream>>>(
                O16, D_N, 0, Wo, D_N, 0, sa_bo + l * D_N, tmpf, D_N, 0, D_N, 1.f);
            k_ln<<<A_N, D_N, 0, stream>>>(xf, tmpf, ln1_g + l * D_N, ln1_b + l * D_N, x16);

            // ---- cross-attention to memory ----
            k_gemm_f16<4><<<dim3(4, 2, 1), 128, 0, stream>>>(
                x16, D_N, 0, caWq16 + (size_t)l * D_N * D_N, D_N, 0, ca_bq + l * D_N,
                Q16, D_N, 0, nullptr, 0, 0, D_N, 0);
            k_gemm_f32<4><<<dim3(MEM_N / 64, 2, H_N), 128, 0, stream>>>(
                Q16, D_N, DH, Kmem16 + (size_t)l * MEM_N * D_N, D_N, DH, nullptr,
                Sf, MEM_N, (long)A_N * MEM_N, DH, ISQ);
            k_softmax<<<dim3(A_N, H_N), 256, 0, stream>>>(Sf, amask, P16, MEM_N);
            k_gemm_f16<2><<<dim3(1, 2, H_N), 128, 0, stream>>>(
                P16, MEM_N, (long)A_N * MEM_N,
                VmemT16 + (size_t)l * D_N * MEM_N, MEM_N, (long)DH * MEM_N, nullptr,
                O16, D_N, DH, nullptr, 0, 0, MEM_N, 0);
            k_gemm_f32<4><<<dim3(4, 2, 1), 128, 0, stream>>>(
                O16, D_N, 0, caWo16 + (size_t)l * D_N * D_N, D_N, 0, ca_bo + l * D_N,
                tmpf, D_N, 0, D_N, 1.f);
            k_ln<<<A_N, D_N, 0, stream>>>(xf, tmpf, ln2_g + l * D_N, ln2_b + l * D_N, x16);

            // ---- FFN ----
            k_gemm_f16<4><<<dim3(MLP_N / 64, 2, 1), 128, 0, stream>>>(
                x16, D_N, 0, ffW1_16 + (size_t)l * MLP_N * D_N, D_N, 0, ff_b1 + l * MLP_N,
                H116, MLP_N, 0, nullptr, 0, 0, D_N, 1 /*relu*/);
            k_gemm_f32<4><<<dim3(4, 2, 1), 128, 0, stream>>>(
                H116, MLP_N, 0, ffW2_16 + (size_t)l * D_N * MLP_N, MLP_N, 0, ff_b2 + l * D_N,
                tmpf, D_N, 0, MLP_N, 1.f);
            k_ln<<<A_N, D_N, 0, stream>>>(xf, tmpf, ln3_g + l * D_N, ln3_b + l * D_N, x16);
        }
        k_outproj<<<1, A_N, 0, stream>>>(xf, out_W, out_b, sp, outp + (size_t)s * A_N * 2);
    }
}
